// TemporalPatternAttention_14276471292385
// MI455X (gfx1250) — matrix operand order repaired and checked by reading, not compiled, not run on hardware
//
#include <hip/hip_runtime.h>

typedef __attribute__((ext_vector_type(16))) __bf16 v16bf;
typedef __attribute__((ext_vector_type(8)))  float  v8f;
typedef __attribute__((ext_vector_type(8)))  unsigned short u16x8;
typedef __attribute__((ext_vector_type(4)))  int    i32x4;

#define HH 8
#define SS 2048
#define DD 512

// ---------- CDNA5 async global->LDS path (guarded; falls back to sync staging) ----------
#if defined(__HIP_DEVICE_COMPILE__) && \
    __has_builtin(__builtin_amdgcn_global_load_async_to_lds_b128) && \
    __has_builtin(__builtin_amdgcn_s_wait_asynccnt)
#define HAVE_ASYNC 1
#define ASYNC_WAIT(n) __builtin_amdgcn_s_wait_asynccnt(n)
#else
#define HAVE_ASYNC 0
#define ASYNC_WAIT(n) ((void)0)
#endif

__device__ __forceinline__ void cp16_g2l(const unsigned short* g, unsigned short* l) {
#if HAVE_ASYNC
  __builtin_amdgcn_global_load_async_to_lds_b128(
      (__attribute__((address_space(1))) i32x4*)g,
      (__attribute__((address_space(3))) i32x4*)l, 0, 0);
#else
  *(u16x8*)l = *(const u16x8*)g;
#endif
}

// ---------- helpers ----------
__device__ __forceinline__ unsigned short f32_bf16(float f) {
  return __builtin_bit_cast(unsigned short, static_cast<__bf16>(f));
}
// A-frag: two contiguous 8-element bf16 runs (16B each, 16B-aligned)
__device__ __forceinline__ v16bf frag_bf_2x8(const unsigned short* p0, const unsigned short* p1) {
  union { u16x8 h[2]; v16bf v; } u;
  u.h[0] = *(const u16x8*)p0;
  u.h[1] = *(const u16x8*)p1;
  return u.v;
}
// B-frag: one contiguous 16-element bf16 run (32B, 32B-aligned)
__device__ __forceinline__ v16bf frag_bf_16(const unsigned short* p) {
  union { u16x8 h[2]; v16bf v; } u;
  u.h[0] = *(const u16x8*)p;
  u.h[1] = *(const u16x8*)(p + 8);
  return u.v;
}
__device__ __forceinline__ v8f wmma_bf16(v16bf a, v16bf b, v8f c) {
  return __builtin_amdgcn_wmma_f32_16x16x32_bf16(false, a, false, b, (short)0, c, false, false);
}

// ---------- one-time f32 -> bf16 conversion (vectorized, grid-stride) ----------
__global__ __launch_bounds__(256) void cvt_kernel(
    const float* __restrict__ in, unsigned short* __restrict__ out, long n) {
  long i = ((long)blockIdx.x * blockDim.x + threadIdx.x) * 4;
  for (; i < n; i += (long)gridDim.x * blockDim.x * 4) {
    float4 v = *(const float4*)(in + i);
    unsigned short s[4];
    s[0] = f32_bf16(v.x); s[1] = f32_bf16(v.y);
    s[2] = f32_bf16(v.z); s[3] = f32_bf16(v.w);
    *(uint2*)(out + i) = *(uint2*)&s[0];
  }
}

// ---------- QKV projection: Y = X @ W^T + b (bf16 in, bf16 out, head-split) ----------
// mode 0: out[((b*8+h)*2048+s)*64+hd]   (Q, K)
// mode 1: out[((b*8+h)*64+hd)*2048+s]   (V transposed for PV B-fragments)
__global__ __launch_bounds__(256) void proj_kernel(
    const unsigned short* __restrict__ X, const unsigned short* __restrict__ W,
    const float* __restrict__ bias, unsigned short* __restrict__ out, int mode) {
  const int lane = threadIdx.x & 31, wave = threadIdx.x >> 5;
  const int hlf = lane >> 4, l16 = lane & 15;
  const int m0 = blockIdx.x * 128 + wave * 16;
  const int n0 = blockIdx.y * 64;
  const unsigned short* xrow = X + (long)(m0 + l16) * DD;

  v8f vz = {0.f, 0.f, 0.f, 0.f, 0.f, 0.f, 0.f, 0.f};
  v8f acc[4];
#pragma unroll
  for (int t = 0; t < 4; ++t) acc[t] = vz;

  for (int k0 = 0; k0 < DD; k0 += 32) {
    v16bf a = frag_bf_2x8(xrow + k0 + 16 * hlf, xrow + k0 + 16 * hlf + 8);
#pragma unroll
    for (int t = 0; t < 4; ++t) {
      const unsigned short* wrow = W + (long)(n0 + 16 * t + l16) * DD;
      v16bf bf = frag_bf_16(wrow + k0 + 16 * hlf);
      acc[t] = wmma_bf16(a, bf, acc[t]);
    }
  }
#pragma unroll
  for (int t = 0; t < 4; ++t) {
#pragma unroll
    for (int j = 0; j < 8; ++j) {
      int m = m0 + j + 8 * hlf;
      int n = n0 + 16 * t + l16;
      float v = acc[t][j] + bias[n];
      int bidx = m >> 11, s_ = m & (SS - 1);
      int h = n >> 6, hd = n & 63;
      long idx = (mode == 0)
          ? ((long)((bidx * HH + h) * SS + s_)) * 64 + hd
          : ((long)((bidx * HH + h) * 64 + hd)) * SS + s_;
      out[idx] = f32_bf16(v);
    }
  }
}

// ---------- flash attention: 8 waves x 16 queries; K/V double-buffered in LDS ----------
__global__ __launch_bounds__(256) void attn_kernel(
    const unsigned short* __restrict__ Q, const unsigned short* __restrict__ K,
    const unsigned short* __restrict__ VT, const float* __restrict__ rel,
    const unsigned char* __restrict__ msk, unsigned short* __restrict__ O) {
  __shared__ float bias_strip[192];
  __shared__ __align__(16) unsigned short Plds[8][16 * 64];
  __shared__ __align__(16) unsigned short Kl[2][64 * 64];  // [key][hd]
  __shared__ __align__(16) unsigned short Vl[2][64 * 64];  // [hd][key]

  const int tid = threadIdx.x;
  const int lane = tid & 31, wave = tid >> 5;
  const int hlf = lane >> 4, l16 = lane & 15;
  const int bh = blockIdx.y;               // b*H + h
  const int b = bh >> 3, h = bh & 7;
  const int q0 = blockIdx.x * 128;
  const int qw = q0 + wave * 16;

  const unsigned short* Kbase = K + (long)bh * SS * 64;    // rows contiguous (stride 64)
  const unsigned short* Vbase = VT + (long)bh * 64 * SS;   // rows stride SS

  // cooperative stage of one 64-key block: K flat 8KB + V 64 rows x 128B
  auto stage_kv = [&](long kb, unsigned short* kl, unsigned short* vl) {
#pragma unroll
    for (int r = 0; r < 2; ++r) {
      int c = tid + 256 * r;               // 0..511 chunks of 16B
      cp16_g2l(Kbase + kb * 64 + c * 8, kl + c * 8);
      int row = c >> 3, off = (c & 7) * 8;
      cp16_g2l(Vbase + kb + (long)row * SS + off, vl + row * 64 + off);
    }
  };

  // persistent Q A-fragments (contraction = head_dim 64 -> 2 chunks of 32)
  const unsigned short* qbase = Q + ((long)bh * SS + qw) * 64;
  v16bf qa[2];
#pragma unroll
  for (int c = 0; c < 2; ++c) {
    const unsigned short* qr = qbase + (long)l16 * 64 + 32 * c + 16 * hlf;
    qa[c] = frag_bf_2x8(qr, qr + 8);
  }

  float rowmax[8], rowsum[8];
  v8f vz = {0.f, 0.f, 0.f, 0.f, 0.f, 0.f, 0.f, 0.f};
  v8f acc[4];
#pragma unroll
  for (int j = 0; j < 8; ++j) { rowmax[j] = -__builtin_inff(); rowsum[j] = 0.f; }
#pragma unroll
  for (int t = 0; t < 4; ++t) acc[t] = vz;

  const float scale = 0.125f;  // 1/sqrt(64)
  unsigned short* pw = &Plds[wave][0];

  stage_kv(0, Kl[0], Vl[0]);   // preload first key block (4 async ops/wave)

  int cur = 0;
  for (int kb = 0; kb < SS; kb += 64, cur ^= 1) {
    __syncthreads();  // everyone done reading both buffers + bias strip
    {                 // stage rel-pos bias strip for this (q-tile, key-block)
      int base = (q0 - kb - 63) + (SS - 1);
      for (int t = tid; t < 191; t += 256)
        bias_strip[t] = rel[(long)(base + t) * HH + h];
    }
    if (kb + 64 < SS) {
      stage_kv(kb + 64, Kl[cur ^ 1], Vl[cur ^ 1]);  // prefetch next block
      ASYNC_WAIT(4);                                // drain current block's 4 ops
    } else {
      ASYNC_WAIT(0);
    }
    // warm L2 two blocks ahead (wrapped, uniform)
    __builtin_prefetch(Kbase + (long)((kb + 128) & (SS - 1)) * 64 + (tid & 127) * 32, 0, 1);
    __syncthreads();  // publish staged K/V + bias strip

    const unsigned short* kl = Kl[cur];
    const unsigned short* vl = Vl[cur];

    // ---- scores: S = (Q @ K^T) * scale + bias + mask ----
    float st[4][8];
#pragma unroll
    for (int t = 0; t < 4; ++t) {
      const unsigned short* kr = kl + (16 * t + l16) * 64;
      v16bf kf0 = frag_bf_16(kr + 16 * hlf);
      v16bf kf1 = frag_bf_16(kr + 32 + 16 * hlf);
      v8f s = vz;
      s = wmma_bf16(qa[0], kf0, s);
      s = wmma_bf16(qa[1], kf1, s);
      float mval = msk[(long)b * SS + kb + 16 * t + l16] ? 0.f : -__builtin_inff();
#pragma unroll
      for (int j = 0; j < 8; ++j) {
        int dp = wave * 16 + j + 8 * hlf - 16 * t - l16 + 63;  // (i-q0)-(key-kb)+63
        st[t][j] = s[j] * scale + bias_strip[dp] + mval;
      }
    }

    // ---- online softmax (rows live across 16 lanes; shfl_xor reductions) ----
    float alpha[8];
#pragma unroll
    for (int j = 0; j < 8; ++j) {
      float bm = fmaxf(fmaxf(st[0][j], st[1][j]), fmaxf(st[2][j], st[3][j]));
#pragma unroll
      for (int off = 1; off < 16; off <<= 1)
        bm = fmaxf(bm, __shfl_xor(bm, off, 32));
      float nm = fmaxf(rowmax[j], bm);
      float al = __expf(rowmax[j] - nm);
      float ps = 0.f;
#pragma unroll
      for (int t = 0; t < 4; ++t) {
        float p = __expf(st[t][j] - nm);
        st[t][j] = p;
        ps += p;
      }
#pragma unroll
      for (int off = 1; off < 16; off <<= 1)
        ps += __shfl_xor(ps, off, 32);
      rowsum[j] = rowsum[j] * al + ps;
      rowmax[j] = nm;
      alpha[j] = al;
    }
#pragma unroll
    for (int t = 0; t < 4; ++t)
#pragma unroll
      for (int j = 0; j < 8; ++j) acc[t][j] *= alpha[j];

    // ---- P (D-layout) -> per-wave LDS tile -> A-fragment layout ----
#pragma unroll
    for (int t = 0; t < 4; ++t)
#pragma unroll
      for (int j = 0; j < 8; ++j)
        pw[(j + 8 * hlf) * 64 + 16 * t + l16] = f32_bf16(st[t][j]);

    v16bf pa[2];
#pragma unroll
    for (int c = 0; c < 2; ++c) {
      const unsigned short* pr = pw + l16 * 64 + 32 * c + 16 * hlf;
      pa[c] = frag_bf_2x8(pr, pr + 8);
    }

    // ---- O += P @ V (V tile already transposed in LDS: [hd][key]) ----
#pragma unroll
    for (int t = 0; t < 4; ++t) {
      const unsigned short* vr = vl + (16 * t + l16) * 64;
      v16bf vf0 = frag_bf_16(vr + 16 * hlf);
      v16bf vf1 = frag_bf_16(vr + 32 + 16 * hlf);
      acc[t] = wmma_bf16(pa[0], vf0, acc[t]);
      acc[t] = wmma_bf16(pa[1], vf1, acc[t]);
    }
  }

  // ---- normalize and store O (bf16, [B,H,S,64]) ----
  float inv[8];
#pragma unroll
  for (int j = 0; j < 8; ++j) inv[j] = 1.f / rowsum[j];
  unsigned short* ob = O + ((long)bh * SS + qw) * 64;
#pragma unroll
  for (int t = 0; t < 4; ++t)
#pragma unroll
    for (int j = 0; j < 8; ++j)
      ob[(j + 8 * hlf) * 64 + 16 * t + l16] = f32_bf16(acc[t][j] * inv[j]);
}

// ---------- output projection: Y = O @ Wo^T + bo (f32 out) ----------
__global__ __launch_bounds__(256) void oproj_kernel(
    const unsigned short* __restrict__ O, const unsigned short* __restrict__ Wo,
    const float* __restrict__ bo, float* __restrict__ Y) {
  const int lane = threadIdx.x & 31, wave = threadIdx.x >> 5;
  const int hlf = lane >> 4, l16 = lane & 15;
  const int m0 = blockIdx.x * 128 + wave * 16;
  const int n0 = blockIdx.y * 64;
  const int m = m0 + l16;
  const int bidx = m >> 11, s_ = m & (SS - 1);

  v8f vz = {0.f, 0.f, 0.f, 0.f, 0.f, 0.f, 0.f, 0.f};
  v8f acc[4];
#pragma unroll
  for (int t = 0; t < 4; ++t) acc[t] = vz;

  for (int k0 = 0; k0 < DD; k0 += 32) {
    int ka = k0 + 16 * hlf;
    int kb = k0 + 16 * hlf + 8;
    const unsigned short* p0 =
        O + ((long)((bidx * HH + (ka >> 6)) * SS + s_)) * 64 + (ka & 63);
    const unsigned short* p1 =
        O + ((long)((bidx * HH + (kb >> 6)) * SS + s_)) * 64 + (kb & 63);
    v16bf a = frag_bf_2x8(p0, p1);
#pragma unroll
    for (int t = 0; t < 4; ++t) {
      const unsigned short* wrow = Wo + (long)(n0 + 16 * t + l16) * DD;
      v16bf bf = frag_bf_16(wrow + k0 + 16 * hlf);
      acc[t] = wmma_bf16(a, bf, acc[t]);
    }
  }
#pragma unroll
  for (int t = 0; t < 4; ++t)
#pragma unroll
    for (int j = 0; j < 8; ++j) {
      int mm = m0 + j + 8 * hlf;
      int n = n0 + 16 * t + l16;
      Y[(long)mm * DD + n] = acc[t][j] + bo[n];
    }
}

// ---------- launch ----------
extern "C" void kernel_launch(void* const* d_in, const int* in_sizes, int n_in,
                              void* d_out, int out_size, void* d_ws, size_t ws_size,
                              hipStream_t stream) {
  const float* x   = (const float*)d_in[0];
  const float* Wq  = (const float*)d_in[1];
  const float* bq  = (const float*)d_in[2];
  const float* Wk  = (const float*)d_in[3];
  const float* bk  = (const float*)d_in[4];
  const float* Wv  = (const float*)d_in[5];
  const float* bv  = (const float*)d_in[6];
  const float* Wo  = (const float*)d_in[7];
  const float* bo  = (const float*)d_in[8];
  const float* rel = (const float*)d_in[9];
  const unsigned char* msk = (const unsigned char*)d_in[10];

  const size_t elems = 4ull * HH * SS * 64;  // 4,194,304 bf16 per [B,H,S,64] tensor
  const size_t welems = (size_t)DD * DD;     // 262,144 per weight matrix
  unsigned short* Qws = (unsigned short*)d_ws;
  unsigned short* Kws = Qws + elems;
  unsigned short* Vws = Kws + elems;
  unsigned short* Ows = Vws + elems;
  unsigned short* Xbf = Ows + elems;
  unsigned short* Wqb = Xbf + elems;
  unsigned short* Wkb = Wqb + welems;
  unsigned short* Wvb = Wkb + welems;
  unsigned short* Wob = Wvb + welems;

  dim3 blk(256);
  cvt_kernel<<<dim3(2048), blk, 0, stream>>>(x,  Xbf, (long)elems);
  cvt_kernel<<<dim3(256),  blk, 0, stream>>>(Wq, Wqb, (long)welems);
  cvt_kernel<<<dim3(256),  blk, 0, stream>>>(Wk, Wkb, (long)welems);
  cvt_kernel<<<dim3(256),  blk, 0, stream>>>(Wv, Wvb, (long)welems);
  cvt_kernel<<<dim3(256),  blk, 0, stream>>>(Wo, Wob, (long)welems);

  dim3 gproj(64, 8);
  proj_kernel<<<gproj, blk, 0, stream>>>(Xbf, Wqb, bq, Qws, 0);
  proj_kernel<<<gproj, blk, 0, stream>>>(Xbf, Wkb, bk, Kws, 0);
  proj_kernel<<<gproj, blk, 0, stream>>>(Xbf, Wvb, bv, Vws, 1);
  attn_kernel<<<dim3(16, 32), blk, 0, stream>>>(Qws, Kws, Vws, rel, msk, Ows);
  oproj_kernel<<<gproj, blk, 0, stream>>>(Ows, Wob, bo, (float*)d_out);
}
